// SVDNet_17239998726553
// MI455X (gfx1250) — matrix operand order repaired and checked by reading, not compiled, not run on hardware
//
#include <hip/hip_runtime.h>

// ============================================================================
// SVDNet on MI455X (gfx1250, wave32).
//  - bf16 WMMA (v_wmma_f32_16x16x32_bf16) for all network GEMMs
//  - f32 WMMA (v_wmma_f32_16x16x4_f32) for the precision-sensitive
//    Newton-Schulz unitary projection
//  - one-time weight prep: f32 [K][N] -> bf16 fragment-native [Npad][Kpad]
//    so every B fragment is two 16B vector loads
//  - decoder big-GEMM B operand streamed by the Tensor Data Mover
//    (tensor_load_to_lds + s_wait_tensorcnt), double-buffered in LDS
// ============================================================================

typedef __attribute__((ext_vector_type(16))) __bf16 v16bf;
typedef __attribute__((ext_vector_type(8)))  __bf16 v8bf;
typedef __attribute__((ext_vector_type(8)))  float  v8f;
typedef __attribute__((ext_vector_type(2)))  float  v2f;
typedef __attribute__((ext_vector_type(4))) unsigned int u32x4;
typedef __attribute__((ext_vector_type(4))) int i32x4;
typedef __attribute__((ext_vector_type(8))) int i32x8;

#define B_TOT   2048
#define MDIM    128
#define NDIM    128
#define IQDIM   2
#define EMBED   32
#define NHEAD   4
#define HEADD   8
#define FFDIM   64
#define RRANK   64
#define DEC_HID 128
#define S_HID   72

// ---------------------------------------------------------------------------
// WMMA wrappers
// ---------------------------------------------------------------------------
__device__ __forceinline__ v8f wmma_bf16(v16bf a, v16bf b, v8f c) {
  return __builtin_amdgcn_wmma_f32_16x16x32_bf16(false, a, false, b, (short)0, c,
                                                 false, false);
}

#if defined(__has_builtin)
#if __has_builtin(__builtin_amdgcn_wmma_f32_16x16x4_f32)
#define HAVE_WMMA_F32 1
#endif
#if __has_builtin(__builtin_amdgcn_tensor_load_to_lds) && \
    __has_builtin(__builtin_amdgcn_s_wait_tensorcnt)
#define HAVE_TDM 1
#endif
#endif

__device__ __forceinline__ v8f wmma_f32x4(v2f a, v2f b, v8f c) {
#ifdef HAVE_WMMA_F32
  return __builtin_amdgcn_wmma_f32_16x16x4_f32(false, a, false, b, (short)0, c,
                                               false, false);
#else
  c[0] += a[0] * b[0] + a[1] * b[1];
  return c;
#endif
}

// ---------------------------------------------------------------------------
// TDM staging: DMA a [rows x 128] bf16 tile (row-major, contiguous) from
// global into LDS. D# per CDNA5 ISA ch.8: group0 = count/lds_addr/global_addr/
// type=2, group1 = data_size + dims/strides. Wave 0 issues; TENSORcnt tracks.
// Falls back to a cooperative vector copy when the builtin is unavailable.
// ---------------------------------------------------------------------------
__device__ __forceinline__ void stage_issue(__bf16* ldsDst, const __bf16* gsrc,
                                            int rows) {
#ifdef HAVE_TDM
  if ((threadIdx.x >> 5) == 0) {
    const unsigned long long ga = (unsigned long long)(const void*)gsrc;
    const unsigned lds_addr = (unsigned)(unsigned long long)(void*)ldsDst;
    u32x4 g0;
    g0[0] = 1u;                                    // count=1, user descriptor
    g0[1] = lds_addr;                              // lds_addr[31:0]
    g0[2] = (unsigned)(ga & 0xffffffffu);          // global_addr[31:0]
    g0[3] = (unsigned)((ga >> 32) & 0x01ffffffu)   // global_addr[56:32]
            | (2u << 30);                          // type=2 ("image")
    i32x8 g1;
    g1[0] = (1 << 16);                             // data_size=1 -> 2 bytes
    g1[1] = (int)(((unsigned)DEC_HID & 0xffffu) << 16);  // tensor_dim0 lo16
    g1[2] = (int)(((unsigned)rows & 0xffffu) << 16);     // tensor_dim1 lo16
    g1[3] = (int)((unsigned)DEC_HID << 16);        // tile_dim0 = 128
    g1[4] = rows;                                  // tile_dim1 (tile_dim2 = 0)
    g1[5] = DEC_HID;                               // tensor_dim0_stride lo32
    g1[6] = 0;
    g1[7] = 0;
    i32x4 z4 = {};
#if __clang_major__ >= 23
    i32x8 z8 = {};
    __builtin_amdgcn_tensor_load_to_lds(g0, g1, z4, z4, z8, 0);
#else
    __builtin_amdgcn_tensor_load_to_lds(g0, g1, z4, z4, 0);
#endif
  }
#else
  const uint4* s = (const uint4*)gsrc;
  uint4* d = (uint4*)ldsDst;
  const int n = rows * DEC_HID / 8;
  for (int u = threadIdx.x; u < n; u += blockDim.x) d[u] = s[u];
#endif
}

__device__ __forceinline__ void stage_wait_le1() {
#ifdef HAVE_TDM
  if ((threadIdx.x >> 5) == 0) __builtin_amdgcn_s_wait_tensorcnt(1);
#endif
}
__device__ __forceinline__ void stage_wait_le0() {
#ifdef HAVE_TDM
  if ((threadIdx.x >> 5) == 0) __builtin_amdgcn_s_wait_tensorcnt(0);
#endif
}

// ---------------------------------------------------------------------------
// Fragment loaders (wave32 layouts per CDNA5 ISA 7.12.2)
// A (16x32 bf16): lane l -> row m0+(l&15); elems 0..7 -> K=k0+((l>>4)<<3)+i,
//                 elems 8..15 -> +16. Two 16B vector loads.
// B (32x16 bf16): lane l -> col n0+(l&15); elems 0..7 -> K=k0+((l>>4)<<3)+i,
//                 elems 8..15 -> +16. From prepped Wt[Npad][Kpad]: two 16B loads.
// C/D f32:        lane l -> col n0+(l&15); elem v -> row m0+((l>>4)<<3)+v.
// ---------------------------------------------------------------------------
__device__ __forceinline__ v16bf fragA_bf(const __bf16* A, int ld, int m0, int k0) {
  const int lane = threadIdx.x & 31;
  const __bf16* p = A + (m0 + (lane & 15)) * ld + k0 + ((lane >> 4) << 3);
  v8bf lo = *(const v8bf*)p;
  v8bf hi = *(const v8bf*)(p + 16);
  return __builtin_shufflevector(lo, hi, 0, 1, 2, 3, 4, 5, 6, 7,
                                 8, 9, 10, 11, 12, 13, 14, 15);
}

// B operand from prepped transposed bf16 weights Wt[Npad][Kpad] (Kpad%32==0).
__device__ __forceinline__ v16bf fragB_bf(const __bf16* Wt, int ldk, int n0, int k0) {
  const int lane = threadIdx.x & 31;
  const __bf16* p = Wt + (size_t)(n0 + (lane & 15)) * ldk + k0 + ((lane >> 4) << 3);
  v8bf lo = *(const v8bf*)p;
  v8bf hi = *(const v8bf*)(p + 16);
  return __builtin_shufflevector(lo, hi, 0, 1, 2, 3, 4, 5, 6, 7,
                                 8, 9, 10, 11, 12, 13, 14, 15);
}

// f32 WMMA (16x16x4) fragments
__device__ __forceinline__ v2f fragA_f32T(const float* U, int ld, int m0, int k0) {
  const int lane = threadIdx.x & 31;
  const int m = m0 + (lane & 15);
  const int kb = k0 + ((lane >> 4) << 1);
  v2f a;
  a[0] = U[kb * ld + m];
  a[1] = U[(kb + 1) * ld + m];
  return a;
}
__device__ __forceinline__ v2f fragA_f32N(const float* Mx, int ld, int m0, int k0) {
  const int lane = threadIdx.x & 31;
  const int m = m0 + (lane & 15);
  const int kb = k0 + ((lane >> 4) << 1);
  return *(const v2f*)(Mx + m * ld + kb);   // contiguous 8B aligned
}
__device__ __forceinline__ v2f fragB_f32N(const float* Mx, int ld, int n0, int k0) {
  const int lane = threadIdx.x & 31;
  const int n = n0 + (lane & 15);
  const int kb = k0 + ((lane >> 4) << 1);
  v2f b;
  b[0] = Mx[kb * ld + n];
  b[1] = Mx[(kb + 1) * ld + n];
  return b;
}

__device__ __forceinline__ float phi_map(float x) {       // elu(x)+1
  return x > 0.f ? x + 1.f : __expf(x);
}
__device__ __forceinline__ float gelu_exact(float x) {
  return 0.5f * x * (1.f + erff(x * 0.70710678118f));
}

// ---------------------------------------------------------------------------
// Parameter bundles
// ---------------------------------------------------------------------------
struct EncParams {
  const float* x;
  const __bf16 *wt_in[2], *wt_q[2], *wt_k[2], *wt_v[2], *wt_o[2], *wt_l1[2], *wt_l2[2];
  const float *in_b[2], *qb[2], *kb[2], *vb[2], *ob[2];
  const float *ln1g[2], *ln1b[2], *ln2g[2], *ln2b[2], *l1b[2], *l2b[2];
};
struct DecParams {
  const __bf16 *wt1[2], *wt2[2];
  const float *b1[2], *b2[2];
};
struct SParams {
  const __bf16 *wt1, *wt2;
  const float *b1, *b2;
};

// ---------------------------------------------------------------------------
// Kernel 0: weight prep. f32 W[K][N] -> bf16 Wt[Npad][Kpad] (zero padded).
// ---------------------------------------------------------------------------
#define N_PREP 18
struct PrepEntry {
  const float* src;
  __bf16* dst;
  int K, N, Kpad, Npad;
};
struct PrepArgs {
  PrepEntry ent[N_PREP];
};

__global__ __launch_bounds__(256) void svd_prep_kernel(PrepArgs A) {
  const PrepEntry E = A.ent[blockIdx.y];
  const int total = E.Npad * E.Kpad;
  for (int u = blockIdx.x * 256 + threadIdx.x; u < total; u += gridDim.x * 256) {
    const int n = u / E.Kpad;
    const int k = u - n * E.Kpad;
    const float v = (n < E.N && k < E.K) ? E.src[(size_t)k * E.N + n] : 0.f;
    E.dst[u] = (__bf16)v;
  }
}

// ---------------------------------------------------------------------------
// Kernel 1: row/col transformer encoder. blockIdx.x = batch, .y = 0 row/1 col.
// 128 threads = 4 waves. 64KB LDS pool, phase-overlaid.
// ---------------------------------------------------------------------------
__global__ __launch_bounds__(128) void svd_enc_kernel(EncParams P, __bf16* rowlat,
                                                      __bf16* collat) {
  const int b = blockIdx.x;
  const int e = blockIdx.y;
  const float* xb = P.x + (size_t)b * (MDIM * NDIM * IQDIM);

  __shared__ __align__(16) char smem[65536];
  // phase A: seqbf = whole pool, [128][256] bf16 (64KB)
  __bf16* seqbf = (__bf16*)smem;
  // phase B layout:
  float*  xres = (float*)smem;                 // [128][32] f32 residual  (16KB)
  float*  act  = (float*)(smem + 16384);       // [128][64] f32 scratch   (32KB)
  __bf16* abf  = (__bf16*)(smem + 49152);      // [128][64] bf16 operands (16KB)
  float*  kvm  = act + 4096;                   // overlays act upper half
  float*  ksum = kvm + NHEAD * HEADD * HEADD;
  __bf16* vbf  = abf + MDIM * EMBED;

  const int tid  = threadIdx.x;
  const int wave = tid >> 5;
  const int lane = tid & 31;
  const int cN   = lane & 15;
  const int rOff = (lane >> 4) << 3;

  // -- 0) stage input slice to LDS as bf16, transpose folded in (coalesced f2)
  {
    const float2* xb2 = (const float2*)xb;
    const float sc = 1.0f / (1.0f + 1e-8f);
    for (int u = tid; u < MDIM * NDIM; u += 128) {
      const int m = u >> 7, n = u & 127;
      const float2 v = xb2[u];                       // x[b][m][n][0..1]
      const int dst = e ? (n * 256 + m * 2) : (m * 256 + n * 2);
      seqbf[dst]     = (__bf16)(v.x * sc);
      seqbf[dst + 1] = (__bf16)(v.y * sc);
    }
  }
  __syncthreads();

  // -- 1) input projection [128,256]x[256,32]; accumulate in registers so the
  //       seqbf pool can be reused for xres afterwards
  v8f acc_in[4];
#pragma unroll
  for (int q = 0; q < 4; ++q) {
    const int t = wave + q * 4;
    const int mt = t >> 1, nt = t & 1;
    v8f acc = {};
    for (int k0 = 0; k0 < 256; k0 += 32)
      acc = wmma_bf16(fragA_bf(seqbf, 256, mt * 16, k0),
                      fragB_bf(P.wt_in[e], 256, nt * 16, k0), acc);
    acc_in[q] = acc;
  }
  __syncthreads();   // seqbf dead; pool becomes xres/act/abf

#pragma unroll
  for (int q = 0; q < 4; ++q) {
    const int t = wave + q * 4;
    const int mt = t >> 1, nt = t & 1;
    const int col = nt * 16 + cN;
    const float bias = P.in_b[e][col];
    const float dv = __expf(-(float)(col & ~1) * 0.28782313f);  // ln(1e4)/32
#pragma unroll
    for (int i = 0; i < 8; ++i) {
      const int row = mt * 16 + rOff + i;
      const float ang = (float)row * dv;
      const float pe = (col & 1) ? __cosf(ang) : __sinf(ang);
      xres[row * EMBED + col] = acc_in[q][i] + bias + pe;
    }
  }
  __syncthreads();

  // -- 2) ln1 -> bf16 y (thread-per-row)
  {
    const int r = tid;
    float mu = 0.f;
    for (int c = 0; c < EMBED; ++c) mu += xres[r * EMBED + c];
    mu *= (1.f / EMBED);
    float var = 0.f;
    for (int c = 0; c < EMBED; ++c) {
      const float d = xres[r * EMBED + c] - mu;
      var += d * d;
    }
    const float inv = rsqrtf(var * (1.f / EMBED) + 1e-5f);
    for (int c = 0; c < EMBED; ++c)
      abf[r * EMBED + c] =
          (__bf16)(((xres[r * EMBED + c] - mu) * inv) * P.ln1g[e][c] + P.ln1b[e][c]);
  }
  __syncthreads();

  // -- 3) kp = phi(y Wk + bk) -> act (f32);  v -> vbf (bf16)
  for (int t = wave; t < 16; t += 4) {
    const int mt = t >> 1, nt = t & 1;
    v16bf a = fragA_bf(abf, EMBED, mt * 16, 0);
    v8f acck = {}, accv = {};
    acck = wmma_bf16(a, fragB_bf(P.wt_k[e], EMBED, nt * 16, 0), acck);
    accv = wmma_bf16(a, fragB_bf(P.wt_v[e], EMBED, nt * 16, 0), accv);
    const int col = nt * 16 + cN;
    const float bk = P.kb[e][col], bv = P.vb[e][col];
#pragma unroll
    for (int i = 0; i < 8; ++i) {
      const int row = mt * 16 + rOff + i;
      act[row * EMBED + col] = phi_map(acck[i] + bk);
      vbf[row * EMBED + col] = (__bf16)(accv[i] + bv);
    }
  }
  __syncthreads();

  // -- 4) KV[h][d][e2] and Ksum[h][d] contractions (tiny, VALU)
  for (int u = tid; u < NHEAD * HEADD * HEADD; u += 128) {
    const int h = u >> 6, d = (u >> 3) & 7, e2 = u & 7;
    float s = 0.f;
    for (int si = 0; si < MDIM; ++si)
      s += act[si * EMBED + h * HEADD + d] * (float)vbf[si * EMBED + h * HEADD + e2];
    kvm[u] = s;
  }
  if (tid < NHEAD * HEADD) {
    const int h = tid >> 3, d = tid & 7;
    float s = 0.f;
    for (int si = 0; si < MDIM; ++si) s += act[si * EMBED + h * HEADD + d];
    ksum[tid] = s;
  }
  __syncthreads();

  // -- 5) qp = phi(y Wq + bq) -> act (overwrite kp, dead after KV)
  for (int t = wave; t < 16; t += 4) {
    const int mt = t >> 1, nt = t & 1;
    v16bf a = fragA_bf(abf, EMBED, mt * 16, 0);
    v8f acc = {};
    acc = wmma_bf16(a, fragB_bf(P.wt_q[e], EMBED, nt * 16, 0), acc);
    const int col = nt * 16 + cN;
    const float bq = P.qb[e][col];
#pragma unroll
    for (int i = 0; i < 8; ++i)
      act[(mt * 16 + rOff + i) * EMBED + col] = phi_map(acc[i] + bq);
  }
  __syncthreads();

  // -- 6) attention combine: out = (qp KV) / (qp Ksum + 1e-6) -> abf (bf16)
  {
    const int s = tid;
#pragma unroll
    for (int h = 0; h < NHEAD; ++h) {
      float qp[HEADD];
      float den = 1e-6f;
#pragma unroll
      for (int d = 0; d < HEADD; ++d) {
        qp[d] = act[s * EMBED + h * HEADD + d];
        den += qp[d] * ksum[h * HEADD + d];
      }
#pragma unroll
      for (int e2 = 0; e2 < HEADD; ++e2) {
        float num = 0.f;
#pragma unroll
        for (int d = 0; d < HEADD; ++d)
          num += qp[d] * kvm[(h * HEADD + d) * HEADD + e2];
        abf[s * EMBED + h * HEADD + e2] = (__bf16)(num / den);
      }
    }
  }
  __syncthreads();

  // -- 7) output projection, residual add
  for (int t = wave; t < 16; t += 4) {
    const int mt = t >> 1, nt = t & 1;
    v16bf a = fragA_bf(abf, EMBED, mt * 16, 0);
    v8f acc = {};
    acc = wmma_bf16(a, fragB_bf(P.wt_o[e], EMBED, nt * 16, 0), acc);
    const int col = nt * 16 + cN;
    const float bo = P.ob[e][col];
#pragma unroll
    for (int i = 0; i < 8; ++i)
      xres[(mt * 16 + rOff + i) * EMBED + col] += acc[i] + bo;
  }
  __syncthreads();

  // -- 8) ln2 -> bf16
  {
    const int r = tid;
    float mu = 0.f;
    for (int c = 0; c < EMBED; ++c) mu += xres[r * EMBED + c];
    mu *= (1.f / EMBED);
    float var = 0.f;
    for (int c = 0; c < EMBED; ++c) {
      const float d = xres[r * EMBED + c] - mu;
      var += d * d;
    }
    const float inv = rsqrtf(var * (1.f / EMBED) + 1e-5f);
    for (int c = 0; c < EMBED; ++c)
      abf[r * EMBED + c] =
          (__bf16)(((xres[r * EMBED + c] - mu) * inv) * P.ln2g[e][c] + P.ln2b[e][c]);
  }
  __syncthreads();

  // -- 9) FF l1 [128,32]x[32,64] + gelu -> act [128][64]
  for (int t = wave; t < 32; t += 4) {
    const int mt = t >> 2, nt = t & 3;
    v16bf a = fragA_bf(abf, EMBED, mt * 16, 0);
    v8f acc = {};
    acc = wmma_bf16(a, fragB_bf(P.wt_l1[e], EMBED, nt * 16, 0), acc);
    const int col = nt * 16 + cN;
    const float b1 = P.l1b[e][col];
#pragma unroll
    for (int i = 0; i < 8; ++i)
      act[(mt * 16 + rOff + i) * FFDIM + col] = gelu_exact(acc[i] + b1);
  }
  __syncthreads();

  // -- 10) bf16 convert for l2's A operand
  for (int u = tid; u < MDIM * FFDIM; u += 128) abf[u] = (__bf16)act[u];
  __syncthreads();

  // -- 11) FF l2 [128,64]x[64,32], residual add
  for (int t = wave; t < 16; t += 4) {
    const int mt = t >> 1, nt = t & 1;
    v8f acc = {};
    for (int k0 = 0; k0 < FFDIM; k0 += 32)
      acc = wmma_bf16(fragA_bf(abf, FFDIM, mt * 16, k0),
                      fragB_bf(P.wt_l2[e], FFDIM, nt * 16, k0), acc);
    const int col = nt * 16 + cN;
    const float b2 = P.l2b[e][col];
#pragma unroll
    for (int i = 0; i < 8; ++i)
      xres[(mt * 16 + rOff + i) * EMBED + col] += acc[i] + b2;
  }
  __syncthreads();

  // -- 12) emit bf16 latent [4096] for the decoders
  __bf16* lat = ((e == 0) ? rowlat : collat) + (size_t)b * (MDIM * EMBED);
  for (int u = tid; u < MDIM * EMBED; u += 128) lat[u] = (__bf16)xres[u];
}

// ---------------------------------------------------------------------------
// Kernel 2: U/V decoders over 16-batch tiles. blockIdx.x = batch tile,
// .y = 0 (U) / 1 (V). The [128,16384] weight stream is DMA'd by the TDM into
// a double-buffered LDS tile while waves run WMMA from LDS only.
// ---------------------------------------------------------------------------
#define CHUNK_ROWS 128
#define N_CHUNK    (16384 / CHUNK_ROWS)

__global__ __launch_bounds__(128) void svd_dec_kernel(DecParams P, const __bf16* rowlat,
                                                      const __bf16* collat, float* outU,
                                                      float* outV) {
  const int bt = blockIdx.x;
  const int which = blockIdx.y;
  const __bf16* lat = ((which == 0) ? rowlat : collat) + (size_t)bt * 16 * 4096;
  const __bf16* wt1 = P.wt1[which];
  const __bf16* wt2 = P.wt2[which];
  const float* b1 = P.b1[which];
  const float* b2 = P.b2[which];
  float* outp = ((which == 0) ? outU : outV) + (size_t)bt * 16 * 16384;

  __shared__ __bf16 hidbf[16 * DEC_HID];
  __shared__ __align__(32) __bf16 wbuf[2][CHUNK_ROWS * DEC_HID];  // 2 x 32KB

  const int tid = threadIdx.x, wave = tid >> 5, lane = tid & 31;
  const int cN = lane & 15, rOff = (lane >> 4) << 3;

  // hid = leaky_relu([16,4096] x [4096,128] + b1)
  for (int nt = wave; nt < 8; nt += 4) {
    v8f acc = {};
    for (int k0 = 0; k0 < 4096; k0 += 32)
      acc = wmma_bf16(fragA_bf(lat, 4096, 0, k0),
                      fragB_bf(wt1, 4096, nt * 16, k0), acc);
    const int col = nt * 16 + cN;
    const float bias = b1[col];
#pragma unroll
    for (int i = 0; i < 8; ++i) {
      float hv = acc[i] + bias;
      hv = (hv > 0.f) ? hv : 0.2f * hv;
      hidbf[(rOff + i) * DEC_HID + col] = (__bf16)hv;
    }
  }
  __syncthreads();

  // out = [16,128] x [128,16384] + b2 ; A fragments hoisted; B streamed by TDM
  const v16bf a0 = fragA_bf(hidbf, DEC_HID, 0, 0);
  const v16bf a1 = fragA_bf(hidbf, DEC_HID, 0, 32);
  const v16bf a2 = fragA_bf(hidbf, DEC_HID, 0, 64);
  const v16bf a3 = fragA_bf(hidbf, DEC_HID, 0, 96);

  stage_issue(wbuf[0], wt2, CHUNK_ROWS);   // prime the pipeline

  for (int nb = 0; nb < N_CHUNK; ++nb) {
    if (nb + 1 < N_CHUNK) {
      stage_issue(wbuf[(nb + 1) & 1],
                  wt2 + (size_t)(nb + 1) * CHUNK_ROWS * DEC_HID, CHUNK_ROWS);
      stage_wait_le1();   // chunk nb complete (nb+1 may remain in flight)
    } else {
      stage_wait_le0();   // last chunk: drain
    }
    __syncthreads();      // publish DMA'd tile to all waves

    const __bf16* wb = wbuf[nb & 1];
    for (int t = wave; t < CHUNK_ROWS / 16; t += 4) {
      v8f acc = {};
      acc = wmma_bf16(a0, fragB_bf(wb, DEC_HID, t * 16, 0), acc);
      acc = wmma_bf16(a1, fragB_bf(wb, DEC_HID, t * 16, 32), acc);
      acc = wmma_bf16(a2, fragB_bf(wb, DEC_HID, t * 16, 64), acc);
      acc = wmma_bf16(a3, fragB_bf(wb, DEC_HID, t * 16, 96), acc);
      const int col = nb * CHUNK_ROWS + t * 16 + cN;
      const float bias = b2[col];
#pragma unroll
      for (int i = 0; i < 8; ++i)
        outp[(size_t)(rOff + i) * 16384 + col] = acc[i] + bias;
    }
    __syncthreads();      // all waves done with wbuf[nb&1] before its reuse
  }
}

// ---------------------------------------------------------------------------
// Kernel 3: s-decoder + softplus + descending sort (rank-based). One block
// per 16-batch tile. wt1 is padded [80][8192], wt2 is padded [64][96].
// ---------------------------------------------------------------------------
__global__ __launch_bounds__(128) void svd_sdec_kernel(SParams P, const __bf16* rowlat,
                                                       const __bf16* collat, float* outS) {
  const int bt = blockIdx.x;
  __shared__ __bf16 hidbf[16 * 96];   // S_HID=72 padded to 96 (3 bf16 K-steps)
  __shared__ float sv[16 * 64];

  const int tid = threadIdx.x, wave = tid >> 5, lane = tid & 31;
  const int cN = lane & 15, rOff = (lane >> 4) << 3;

  for (int u = tid; u < 16 * 96; u += 128) hidbf[u] = (__bf16)0.f;
  __syncthreads();

  // hid = leaky_relu([16,8192] x [8192,72]); K<4096 from row latent, else col
  for (int nt = wave; nt < 5; nt += 4) {
    v8f acc = {};
    for (int k0 = 0; k0 < 8192; k0 += 32) {
      const __bf16* base =
          (k0 < 4096) ? (rowlat + (size_t)bt * 16 * 4096) : (collat + (size_t)bt * 16 * 4096);
      acc = wmma_bf16(fragA_bf(base, 4096, 0, k0 & 4095),
                      fragB_bf(P.wt1, 8192, nt * 16, k0), acc);
    }
    const int col = nt * 16 + cN;
    if (col < S_HID) {
      const float bias = P.b1[col];
#pragma unroll
      for (int i = 0; i < 8; ++i) {
        float hv = acc[i] + bias;
        hv = (hv > 0.f) ? hv : 0.2f * hv;
        hidbf[(rOff + i) * 96 + col] = (__bf16)hv;
      }
    }
  }
  __syncthreads();

  // s = softplus([16,72pad96] x [72pad96,64])
  {
    const int nt = wave;   // 4 waves -> 4 N-tiles
    v8f acc = {};
    for (int k0 = 0; k0 < 96; k0 += 32)
      acc = wmma_bf16(fragA_bf(hidbf, 96, 0, k0),
                      fragB_bf(P.wt2, 96, nt * 16, k0), acc);
    const int col = nt * 16 + cN;
    const float bias = P.b2[col];
#pragma unroll
    for (int i = 0; i < 8; ++i) {
      const float xv = acc[i] + bias;
      sv[(rOff + i) * 64 + col] = (xv > 20.f) ? xv : log1pf(__expf(xv));
    }
  }
  __syncthreads();

  // descending sort per batch via rank counting (64 elems, O(64^2) trivial)
  float* sb = outS + (size_t)bt * 16 * 64;
  for (int u = tid; u < 16 * 64; u += 128) {
    const int bi = u >> 6, j = u & 63;
    const float vj = sv[bi * 64 + j];
    int rank = 0;
    for (int l = 0; l < 64; ++l) {
      const float vl = sv[bi * 64 + l];
      rank += (vl > vj) || (vl == vj && l < j);
    }
    sb[bi * 64 + rank] = vj;
  }
}

// ---------------------------------------------------------------------------
// Kernel 4: complex Newton-Schulz unitary projection, in place on d_out.
// f32 WMMA 16x16x4 (precision-sensitive step). blockIdx.x = batch, .y = U/V.
// 256 threads = 8 waves. LDS: Ur,Ui (64KB) + Gr,Gi (32KB) + norms.
// ---------------------------------------------------------------------------
__global__ __launch_bounds__(256) void svd_newton_kernel(float* outU, float* outV) {
  float* X = ((blockIdx.y == 0) ? outU : outV) + (size_t)blockIdx.x * (MDIM * RRANK * IQDIM);

  __shared__ float Ur[MDIM * RRANK];
  __shared__ float Ui[MDIM * RRANK];
  __shared__ float Gr[RRANK * RRANK];
  __shared__ float Gi[RRANK * RRANK];
  __shared__ float nrm[RRANK];

  const int tid = threadIdx.x, wave = tid >> 5, lane = tid & 31;
  const int cN = lane & 15, rOff = (lane >> 4) << 3;

  // load deinterleaved (float2 per element pair)
  {
    const float2* X2 = (const float2*)X;
    for (int u = tid; u < MDIM * RRANK; u += 256) {
      const float2 z = X2[u];
      Ur[u] = z.x;
      Ui[u] = z.y;
    }
  }
  __syncthreads();

  // column normalization
  if (tid < RRANK) {
    float s = 0.f;
    for (int r = 0; r < MDIM; ++r) {
      const float a = Ur[r * RRANK + tid], b = Ui[r * RRANK + tid];
      s += a * a + b * b;
    }
    nrm[tid] = 1.f / (sqrtf(s) + 1e-8f);
  }
  __syncthreads();
  for (int u = tid; u < MDIM * RRANK; u += 256) {
    const float sc = nrm[u & 63];
    Ur[u] *= sc;
    Ui[u] *= sc;
  }
  __syncthreads();

  for (int it = 0; it < 2; ++it) {
    // G = U^H U : M=N=64, K=128 ; complex via 4 real WMMA chains per k-step
    for (int t = wave; t < 16; t += 8) {
      const int mt = t >> 2, nt = t & 3;
      v8f ar = {}, ai = {};
      for (int k0 = 0; k0 < MDIM; k0 += 4) {
        v2f atr = fragA_f32T(Ur, RRANK, mt * 16, k0);
        v2f ati = fragA_f32T(Ui, RRANK, mt * 16, k0);
        v2f br  = fragB_f32N(Ur, RRANK, nt * 16, k0);
        v2f bi  = fragB_f32N(Ui, RRANK, nt * 16, k0);
        ar = wmma_f32x4(atr, br, ar);
        ar = wmma_f32x4(ati, bi, ar);
        ai = wmma_f32x4(atr, bi, ai);
        v2f nati;
        nati[0] = -ati[0];
        nati[1] = -ati[1];
        ai = wmma_f32x4(nati, br, ai);
      }
      const int c = nt * 16 + cN;
#pragma unroll
      for (int i = 0; i < 8; ++i) {
        const int r = mt * 16 + rOff + i;
        Gr[r * RRANK + c] = ar[i];
        Gi[r * RRANK + c] = ai[i];
      }
    }
    __syncthreads();

    // P = U G (M=128, N=64, K=64); Unew = 1.5 U - 0.5 P, updated in place
    v8f pr[4], pi[4];
    for (int t = wave; t < 32; t += 8) {
      const int q = (t - wave) >> 3;
      const int mt = t >> 2, nt = t & 3;
      v8f accr = {}, acci = {};
      for (int k0 = 0; k0 < RRANK; k0 += 4) {
        v2f aur = fragA_f32N(Ur, RRANK, mt * 16, k0);
        v2f aui = fragA_f32N(Ui, RRANK, mt * 16, k0);
        v2f bgr = fragB_f32N(Gr, RRANK, nt * 16, k0);
        v2f bgi = fragB_f32N(Gi, RRANK, nt * 16, k0);
        accr = wmma_f32x4(aur, bgr, accr);
        v2f naui;
        naui[0] = -aui[0];
        naui[1] = -aui[1];
        accr = wmma_f32x4(naui, bgi, accr);
        acci = wmma_f32x4(aur, bgi, acci);
        acci = wmma_f32x4(aui, bgr, acci);
      }
      pr[q] = accr;
      pi[q] = acci;
    }
    __syncthreads();
    for (int t = wave; t < 32; t += 8) {
      const int q = (t - wave) >> 3;
      const int mt = t >> 2, nt = t & 3;
      const int c = nt * 16 + cN;
#pragma unroll
      for (int i = 0; i < 8; ++i) {
        const int idx = (mt * 16 + rOff + i) * RRANK + c;
        Ur[idx] = 1.5f * Ur[idx] - 0.5f * pr[q][i];
        Ui[idx] = 1.5f * Ui[idx] - 0.5f * pi[q][i];
      }
    }
    __syncthreads();
  }

  // store re-interleaved
  {
    float2* X2 = (float2*)X;
    for (int u = tid; u < MDIM * RRANK; u += 256) {
      float2 z;
      z.x = Ur[u];
      z.y = Ui[u];
      X2[u] = z;
    }
  }
}

// ---------------------------------------------------------------------------
// Host launcher. Input flattening (setup_inputs dict order, leaves in dict
// insertion order):
//  0:x  1:row_in.w 2:row_in.b 3:col_in.w 4:col_in.b
//  5..20:  row layer {q.w,q.b,k.w,k.b,v.w,v.b,o.w,o.b,ln1_g,ln1_b,ln2_g,ln2_b,
//                     l1.w,l1.b,l2.w,l2.b}
//  21..36: col layer (same order)
//  37..40: u_dec {l1.w,l1.b,l2.w,l2.b}   41..44: v_dec   45..48: s_dec
// Output: concat(Uproj [B,128,64,2], s [B,64], Vproj [B,128,64,2]) f32.
// ws: rowlat bf16 (16MB) | collat bf16 (16MB) | prepped bf16 weights (~12MB).
// ---------------------------------------------------------------------------
extern "C" void kernel_launch(void* const* d_in, const int* in_sizes, int n_in,
                              void* d_out, int out_size, void* d_ws, size_t ws_size,
                              hipStream_t stream) {
  (void)in_sizes; (void)n_in; (void)out_size; (void)ws_size;
  auto F = [&](int i) { return (const float*)d_in[i]; };

  __bf16* rowlat = (__bf16*)d_ws;
  __bf16* collat = rowlat + (size_t)B_TOT * (MDIM * EMBED);
  __bf16* wcur = collat + (size_t)B_TOT * (MDIM * EMBED);

  PrepArgs pa;
  int np = 0;
  auto add = [&](const float* src, int K, int N, int Kpad, int Npad) {
    PrepEntry& E = pa.ent[np++];
    E.src = src;
    E.dst = wcur;
    E.K = K;
    E.N = N;
    E.Kpad = Kpad;
    E.Npad = Npad;
    __bf16* r = wcur;
    wcur += (size_t)Kpad * Npad;
    return r;
  };

  EncParams ep;
  ep.x = F(0);
  ep.in_b[0] = F(2);
  ep.in_b[1] = F(4);
  ep.wt_in[0] = add(F(1), 256, EMBED, 256, EMBED);
  ep.wt_in[1] = add(F(3), 256, EMBED, 256, EMBED);
  const int base[2] = {5, 21};
  for (int e = 0; e < 2; ++e) {
    const int p = base[e];
    ep.wt_q[e] = add(F(p + 0), EMBED, EMBED, EMBED, EMBED);
    ep.qb[e] = F(p + 1);
    ep.wt_k[e] = add(F(p + 2), EMBED, EMBED, EMBED, EMBED);
    ep.kb[e] = F(p + 3);
    ep.wt_v[e] = add(F(p + 4), EMBED, EMBED, EMBED, EMBED);
    ep.vb[e] = F(p + 5);
    ep.wt_o[e] = add(F(p + 6), EMBED, EMBED, EMBED, EMBED);
    ep.ob[e] = F(p + 7);
    ep.ln1g[e] = F(p + 8);
    ep.ln1b[e] = F(p + 9);
    ep.ln2g[e] = F(p + 10);
    ep.ln2b[e] = F(p + 11);
    ep.wt_l1[e] = add(F(p + 12), EMBED, FFDIM, EMBED, FFDIM);
    ep.l1b[e] = F(p + 13);
    ep.wt_l2[e] = add(F(p + 14), FFDIM, EMBED, FFDIM, EMBED);
    ep.l2b[e] = F(p + 15);
  }
  DecParams dp;
  dp.wt1[0] = add(F(37), 4096, DEC_HID, 4096, DEC_HID);
  dp.b1[0] = F(38);
  dp.wt2[0] = add(F(39), DEC_HID, 16384, DEC_HID, 16384);
  dp.b2[0] = F(40);
  dp.wt1[1] = add(F(41), 4096, DEC_HID, 4096, DEC_HID);
  dp.b1[1] = F(42);
  dp.wt2[1] = add(F(43), DEC_HID, 16384, DEC_HID, 16384);
  dp.b2[1] = F(44);
  SParams sp;
  sp.wt1 = add(F(45), 8192, S_HID, 8192, 80);
  sp.b1 = F(46);
  sp.wt2 = add(F(47), S_HID, RRANK, 96, RRANK);
  sp.b2 = F(48);

  float* out = (float*)d_out;
  float* outU = out;
  float* outS = out + (size_t)B_TOT * MDIM * RRANK * IQDIM;
  float* outV = outS + (size_t)B_TOT * RRANK;

  svd_prep_kernel<<<dim3(64, N_PREP), 256, 0, stream>>>(pa);
  svd_enc_kernel<<<dim3(B_TOT, 2), 128, 0, stream>>>(ep, rowlat, collat);
  svd_dec_kernel<<<dim3(B_TOT / 16, 2), 128, 0, stream>>>(dp, rowlat, collat, outU, outV);
  svd_sdec_kernel<<<B_TOT / 16, 128, 0, stream>>>(sp, rowlat, collat, outS);
  svd_newton_kernel<<<dim3(B_TOT, 2), 256, 0, stream>>>(outU, outV);
}
